// MyModel_87522843559014
// MI455X (gfx1250) — matrix operand order repaired and checked by reading, not compiled, not run on hardware
//
#include <hip/hip_runtime.h>

// ---------------------------------------------------------------------------
// LSTM (B=256, T=512, F=128, H=256) + MLP head on CDNA5 / gfx1250.
//
// One workgroup per 16-row batch tile: 16 waves (one per 16-col hidden tile).
//  - h double-buffered in LDS; per-step sync = split workgroup barrier
//    (s_barrier_signal -1 / s_barrier_wait -1); x-projection overlaps barrier.
//  - rkernel B-fragments register-resident (8 kb x 4 gates x v16bf / wave).
//  - kernel (x-projection) B-fragments resident in LDS (256KB), filled once
//    by the Tensor Data Mover (tensor_load_to_lds + s_wait_tensorcnt).
//  - bf16 WMMA (v_wmma_f32_16x16x32_bf16) with f32 accumulation.
//  - t-loop unroll disabled + bounded temps to stay under the ~512 VGPR/wave
//    budget (4 waves/SIMD) and avoid scratch spills.
// ---------------------------------------------------------------------------

typedef __attribute__((ext_vector_type(16))) __bf16 v16bf;
typedef __attribute__((ext_vector_type(8)))  __bf16 v8bf;
typedef __attribute__((ext_vector_type(8)))  float  v8f;
typedef __attribute__((ext_vector_type(4)))  unsigned int u32x4;
typedef __attribute__((ext_vector_type(8)))  int i32x8;
typedef __attribute__((ext_vector_type(4)))  int i32x4;

#define B_ 256
#define T_ 512
#define F_ 128
#define H_ 256
#define HP 264                      // padded LDS pitch for h rows (bank spread)
#define HSH_BYTES (2 * 16 * HP * 2) // h double buffer: 16896 B (32B aligned)
#define KB_BYTES  (64 * 4 * 512 * 2)       // packed x-projection B: 262144 B
#define SMEM_BYTES (HSH_BYTES + KB_BYTES)  // 279040 B <= 320KB WGP LDS

__device__ __forceinline__ float sigf(float x)     { return 1.0f / (1.0f + __expf(-x)); }
__device__ __forceinline__ float tanhfast(float x) { return 2.0f / (1.0f + __expf(-2.0f * x)) - 1.0f; }

// ---- pack weights into WMMA-B lane layout (bf16) --------------------------
// B-fragment lane map: lane n -> N = (n&15), K = ((n>>4)<<3) + (j&7) + ((j>>3)<<4), j = 0..15.
__global__ void pack_weights(const float* __restrict__ kern,
                             const float* __restrict__ rkern,
                             __bf16* __restrict__ pwk,
                             __bf16* __restrict__ pwr) {
    int idx = blockIdx.x * blockDim.x + threadIdx.x;
    if (idx < 64 * 4 * 512) {                       // 131072: x-projection part
        int j    = idx & 15;
        int lane = (idx >> 4) & 31;
        int rest = idx >> 9;                        // col16*4 + kb
        int kb   = rest & 3;
        int c16  = rest >> 2;
        int col  = c16 * 16 + (lane & 15);
        int K    = kb * 32 + ((lane >> 4) << 3) + (j & 7) + ((j >> 3) << 4);
        pwk[idx] = (__bf16)kern[(size_t)K * (4 * H_) + col];
    } else if (idx < 64 * 4 * 512 + 64 * 8 * 512) { // 262144: recurrent part
        int i2   = idx - 64 * 4 * 512;
        int j    = i2 & 15;
        int lane = (i2 >> 4) & 31;
        int rest = i2 >> 9;                         // col16*8 + kb
        int kb   = rest & 7;
        int c16  = rest >> 3;
        int col  = c16 * 16 + (lane & 15);
        int K    = kb * 32 + ((lane >> 4) << 3) + (j & 7) + ((j >> 3) << 4);
        pwr[i2]  = (__bf16)rkern[(size_t)K * (4 * H_) + col];
    }
}

// ---- persistent LSTM: 16 blocks (one per batch tile) x 512 threads --------
__global__ __launch_bounds__(512) void lstm_persistent(
        const float* __restrict__ x,      // [B, T, F] f32
        const __bf16* __restrict__ pwk,   // packed x-projection B (TDM source)
        const __bf16* __restrict__ pwr,   // packed recurrent B
        float* __restrict__ hlast) {      // [B, H] f32
    extern __shared__ char smem[];
    __bf16* hsh = (__bf16*)smem;                 // [2][16][HP]
    __bf16* kB  = (__bf16*)(smem + HSH_BYTES);   // [64*4*512]

    const int lane  = threadIdx.x & 31;
    const int htile = threadIdx.x >> 5;          // 0..15 (hidden-col tile)
    const int btile = blockIdx.x;                // 0..15 (batch-row tile)
    const int mrow  = lane & 15;
    const int ksel  = (lane >> 4) << 3;          // 0 / 8 (A & D lane-group shift)

    // zero h double buffer
    for (int i = threadIdx.x; i < 2 * 16 * HP; i += 512) hsh[i] = (__bf16)0.0f;

    // ---- fill LDS-resident x-projection B tiles via Tensor Data Mover ----
    if (threadIdx.x == 0) {
        unsigned lds_off = (unsigned)(size_t)(void*)kB;
        unsigned long long ga = (unsigned long long)(size_t)pwk;
        u32x4 g0;
        g0[0] = 1u;                                    // count=1, user descriptor
        g0[1] = lds_off;                               // lds_addr
        g0[2] = (unsigned)(ga & 0xFFFFFFFFu);          // global_addr[31:0]
        g0[3] = (unsigned)((ga >> 32) & 0x01FFFFFFu) | (2u << 30); // ga[56:32] | type=2
        i32x8 g1;                                      // 262144 B as 8B elems: 4096 x 8
        g1[0] = (3 << 16);                             // data_size = 8B
        g1[1] = (int)(4096u << 16);                    // tensor_dim0[15:0]
        g1[2] = (int)(8u << 16);                       // tensor_dim1[15:0]
        g1[3] = (int)(4096u << 16);                    // tile_dim0
        g1[4] = 8;                                     // tile_dim1
        g1[5] = 4096;                                  // tensor_dim0_stride[31:0]
        g1[6] = 0;
        g1[7] = 0;
        i32x4 z4 = {0, 0, 0, 0};
        i32x8 z8 = {0, 0, 0, 0, 0, 0, 0, 0};
        __builtin_amdgcn_tensor_load_to_lds(g0, g1, z4, z4, z8, 0);
        __builtin_amdgcn_s_wait_tensorcnt(0);
    }
    __syncthreads();

    // ---- register-resident recurrent B-fragments: 8 kb x 4 gates ----
    v16bf wR[8][4];
#pragma unroll
    for (int kb = 0; kb < 8; ++kb) {
#pragma unroll
        for (int g = 0; g < 4; ++g) {
            wR[kb][g] = *(const v16bf*)(pwr + ((((size_t)(g * 16 + htile) * 8 + kb) * 32 + lane) << 4));
        }
    }

    v8f zero8 = {0.f, 0.f, 0.f, 0.f, 0.f, 0.f, 0.f, 0.f};
    v8f cst = zero8;                                 // persistent cell state

    const float* xrow0 = x + ((size_t)(btile * 16 + mrow) * T_) * F_;
    const __bf16* kBh  = kB + (((size_t)htile * 4) * 32 + lane) * 16; // htile base

#pragma clang loop unroll(disable)
    for (int t = 0; t < T_; ++t) {
        const __bf16* hread  = hsh + (size_t)((t - 1) & 1) * 16 * HP;
        __bf16*       hwrite = hsh + (size_t)(t & 1) * 16 * HP;

        v8f acc[4] = {zero8, zero8, zero8, zero8};

        // ================= x-projection (independent of h) =================
        const float* xrow = xrow0 + (size_t)t * F_;
#pragma unroll
        for (int kb = 0; kb < 4; ++kb) {
            v16bf aX;
            const float* p = xrow + kb * 32 + ksel;
#pragma unroll
            for (int e = 0; e < 8; ++e) { aX[e] = (__bf16)p[e]; aX[e + 8] = (__bf16)p[e + 16]; }
            v16bf bT0 = *(const v16bf*)(kBh + (0 * 16 * 4 + kb) * 512);
            v16bf bT1 = *(const v16bf*)(kBh + (1 * 16 * 4 + kb) * 512);
            v16bf bT2 = *(const v16bf*)(kBh + (2 * 16 * 4 + kb) * 512);
            v16bf bT3 = *(const v16bf*)(kBh + (3 * 16 * 4 + kb) * 512);
            acc[0] = __builtin_amdgcn_wmma_f32_16x16x32_bf16(false, aX, false, bT0, (short)0, acc[0], false, false);
            acc[1] = __builtin_amdgcn_wmma_f32_16x16x32_bf16(false, aX, false, bT1, (short)0, acc[1], false, false);
            acc[2] = __builtin_amdgcn_wmma_f32_16x16x32_bf16(false, aX, false, bT2, (short)0, acc[2], false, false);
            acc[3] = __builtin_amdgcn_wmma_f32_16x16x32_bf16(false, aX, false, bT3, (short)0, acc[3], false, false);
        }

        if (t + 1 < T_) __builtin_prefetch(xrow + F_, 0, 0);  // next x_t row

        // ---- wait half of split barrier: h(t-1) writes now visible ----
        if (t > 0) asm volatile("s_barrier_wait -1" ::: "memory");

        // ================= recurrence: A from LDS h, B resident ============
        const __bf16* hrow = hread + (size_t)mrow * HP;
        v16bf aH;
        {
            const __bf16* p = hrow + ksel;
            v8bf lo = *(const v8bf*)p;
            v8bf hi = *(const v8bf*)(p + 16);
#pragma unroll
            for (int e = 0; e < 8; ++e) { aH[e] = lo[e]; aH[e + 8] = hi[e]; }
        }
#pragma unroll
        for (int kb = 0; kb < 7; ++kb) {
            v16bf aN;
            const __bf16* p = hrow + (kb + 1) * 32 + ksel;
            v8bf lo = *(const v8bf*)p;
            v8bf hi = *(const v8bf*)(p + 16);
#pragma unroll
            for (int e = 0; e < 8; ++e) { aN[e] = lo[e]; aN[e + 8] = hi[e]; }
            acc[0] = __builtin_amdgcn_wmma_f32_16x16x32_bf16(false, aH, false, wR[kb][0], (short)0, acc[0], false, false);
            acc[1] = __builtin_amdgcn_wmma_f32_16x16x32_bf16(false, aH, false, wR[kb][1], (short)0, acc[1], false, false);
            acc[2] = __builtin_amdgcn_wmma_f32_16x16x32_bf16(false, aH, false, wR[kb][2], (short)0, acc[2], false, false);
            acc[3] = __builtin_amdgcn_wmma_f32_16x16x32_bf16(false, aH, false, wR[kb][3], (short)0, acc[3], false, false);
            aH = aN;
        }
        acc[0] = __builtin_amdgcn_wmma_f32_16x16x32_bf16(false, aH, false, wR[7][0], (short)0, acc[0], false, false);
        acc[1] = __builtin_amdgcn_wmma_f32_16x16x32_bf16(false, aH, false, wR[7][1], (short)0, acc[1], false, false);
        acc[2] = __builtin_amdgcn_wmma_f32_16x16x32_bf16(false, aH, false, wR[7][2], (short)0, acc[2], false, false);
        acc[3] = __builtin_amdgcn_wmma_f32_16x16x32_bf16(false, aH, false, wR[7][3], (short)0, acc[3], false, false);

        // ================= gates + h/c update ==============================
        const int colc = htile * 16 + mrow;
#pragma unroll
        for (int e = 0; e < 8; ++e) {
            float iv = acc[0][e], fv = acc[1][e], gv = acc[2][e], ov = acc[3][e];
            float cn = sigf(fv) * cst[e] + sigf(iv) * tanhfast(gv);
            float hn = sigf(ov) * tanhfast(cn);
            cst[e] = cn;
            int row = e + ksel;                       // D layout: M=e / e+8
            hwrite[(size_t)row * HP + colc] = (__bf16)hn;
            if (t == T_ - 1) hlast[(size_t)(btile * 16 + row) * H_ + colc] = hn;
        }

        // ---- signal half of split barrier (h(t) published) ----
        if (t + 1 < T_) {
            asm volatile("s_wait_dscnt 0x0" ::: "memory");   // LDS stores landed
            asm volatile("s_barrier_signal -1" ::: "memory");
        }
    }
}

// ---- MLP head: relu(h @ w1 + b1) @ w2 + b2  (negligible cost) -------------
__global__ void head_kernel(const float* __restrict__ hlast,
                            const float* __restrict__ w1, const float* __restrict__ b1,
                            const float* __restrict__ w2, const float* __restrict__ b2,
                            float* __restrict__ out) {
    int b = blockIdx.x * blockDim.x + threadIdx.x;
    if (b >= B_) return;
    const float* hr = hlast + (size_t)b * H_;
    float accum = 0.0f;
    for (int j = 0; j < 100; ++j) {
        float s = b1[j];
        for (int k = 0; k < H_; ++k) s = fmaf(hr[k], w1[k * 100 + j], s);
        accum = fmaf(fmaxf(s, 0.0f), w2[j], accum);
    }
    out[b] = accum + b2[0];
}

// ---------------------------------------------------------------------------
extern "C" void kernel_launch(void* const* d_in, const int* in_sizes, int n_in,
                              void* d_out, int out_size, void* d_ws, size_t ws_size,
                              hipStream_t stream) {
    const float* x     = (const float*)d_in[0];
    const float* kern  = (const float*)d_in[1];
    const float* rkern = (const float*)d_in[2];
    const float* w1    = (const float*)d_in[3];
    const float* b1    = (const float*)d_in[4];
    const float* w2    = (const float*)d_in[5];
    const float* b2    = (const float*)d_in[6];
    float* out = (float*)d_out;

    char* ws = (char*)d_ws;
    __bf16* pwk   = (__bf16*)(ws);            // 262144 B
    __bf16* pwr   = (__bf16*)(ws + 262144);   // 524288 B
    float*  hlast = (float*)(ws + 786432);    // 262144 B

    pack_weights<<<1536, 256, 0, stream>>>(kern, rkern, pwk, pwr);
    lstm_persistent<<<16, 512, SMEM_BYTES, stream>>>(x, pwk, pwr, hlast);
    head_kernel<<<8, 32, 0, stream>>>(hlast, w1, b1, w2, b2, out);
}
